// ConditionalARGDenoising_67997922230995
// MI455X (gfx1250) — compile-verified, matrix operand order checked by reading, not run on hardware
//
#include <hip/hip_runtime.h>
#include <hip/hip_bf16.h>

// ---------------- problem constants ----------------
#define N_NODES 20000
#define N_EDGES 320000
#define HID     256
#define NLAYERS 5
#define IN_DIM  112     // PH*NF = 16*7
#define IN_PAD  128
#define NGROUP  64
#define M_CAT   769     // 2*HID + 1 + HID
#define M_PAD   800     // padded to multiple of 32
#define EPS_F   1e-8f
#define GS_NODES 64

typedef __bf16 bf16;
typedef __attribute__((ext_vector_type(16))) __bf16 bf16x16;
typedef __attribute__((ext_vector_type(8)))  __bf16 bf16x8;
typedef __attribute__((ext_vector_type(8)))  float  f32x8;

__device__ __forceinline__ float silu_f(float v) { return v / (1.0f + __expf(-v)); }

// ---------------- CDNA5 async memory->LDS copy (ASYNCcnt tracked) ----------------
// Each lane moves 16 bytes global -> LDS without touching VGPд data registers.
// VDST = per-lane LDS byte offset, VADDR = per-lane 64-bit global address.
__device__ __forceinline__ void async_copy_b128(unsigned lds_off, unsigned long long gaddr) {
  asm volatile("global_load_async_to_lds_b128 %0, %1, off"
               :: "v"(lds_off), "v"(gaddr)
               : "memory");
}
__device__ __forceinline__ void wait_asynccnt0() {
  asm volatile("s_wait_asynccnt 0x0" ::: "memory");
}
__device__ __forceinline__ unsigned lds_addr32(const void* p) {
  return (unsigned)(unsigned long long)(uintptr_t)p;   // low 32 bits = LDS offset
}

// ---------------- WMMA 16x16x32 bf16 fragment helpers (wave32) ----------------
// A (16x32, 16-bit): lanes 0-15 row M=lane hold K=0..7 and K=16..23;
// lanes 16-31 same rows hold K=8..15 and K=24..31.
__device__ __forceinline__ bf16x16 load_a_frag(const bf16* aBase, int lda, int k0) {
  int lane = threadIdx.x & 31;
  int row = lane & 15, kh = lane >> 4;
  const bf16* p = aBase + row * lda + k0 + kh * 8;
  bf16x8 lo = *(const bf16x8*)p;
  bf16x8 hi = *(const bf16x8*)(p + 16);
  bf16x16 r;
#pragma unroll
  for (int i = 0; i < 8; ++i) { r[i] = lo[i]; r[i + 8] = hi[i]; }
  return r;
}

// B fragments pre-packed so each lane reads one contiguous 32-byte chunk.
__device__ __forceinline__ bf16x16 load_b_frag(const bf16* Wp, int ntn, int kt, int nt) {
  int lane = threadIdx.x & 31;
  const bf16* p = Wp + (((size_t)kt * ntn + nt) * 32 + lane) * 16;
  return *(const bf16x16*)p;
}

__device__ __forceinline__ f32x8 wmma_row(const bf16* aBase, int lda,
                                          const bf16* Wp, int ntn, int nt, int kTiles) {
  f32x8 acc = {};
  for (int kt = 0; kt < kTiles; ++kt) {
    bf16x16 a = load_a_frag(aBase, lda, kt * 32);
    bf16x16 b = load_b_frag(Wp, ntn, kt, nt);
    acc = __builtin_amdgcn_wmma_f32_16x16x32_bf16(false, a, false, b, (short)0, acc,
                                                  false, false);
  }
  return acc;
}

// ---------------- weight repack: f32 [K x N] -> packed bf16 fragments ----------------
__global__ void k_repack(const float* __restrict__ W, bf16* __restrict__ Wp,
                         int Ksrc, int Kpad, int Nn) {
  size_t idx = (size_t)blockIdx.x * blockDim.x + threadIdx.x;
  size_t total = (size_t)Kpad * Nn;
  if (idx >= total) return;
  int e = (int)(idx & 15);
  size_t t = idx >> 4;
  int lane = (int)(t & 31); t >>= 5;
  int ntn = Nn >> 4;
  int nt = (int)(t % ntn);
  int kt = (int)(t / ntn);
  int col = lane & 15, kh = lane >> 4;
  int kk = kh * 8 + e + ((e >= 8) ? 8 : 0);
  int k = kt * 32 + kk, n = nt * 16 + col;
  float v = (k < Ksrc) ? W[(size_t)k * Nn + n] : 0.0f;
  Wp[idx] = (bf16)v;
}

// ---------------- node embedding: h = x(N x 112) @ node_w + node_b ----------------
__global__ __launch_bounds__(256) void k_node_embed(const float* __restrict__ x,
                                                    const bf16* __restrict__ Wp,
                                                    const float* __restrict__ bias,
                                                    float* __restrict__ h) {
  __shared__ __align__(16) bf16 a[16 * IN_PAD];
  int tile = blockIdx.x, tid = threadIdx.x;
  for (int i = tid; i < 16 * IN_PAD; i += 256) {
    int r = i >> 7, c = i & 127;
    float v = (c < IN_DIM) ? x[(size_t)(tile * 16 + r) * IN_DIM + c] : 0.0f;
    a[i] = (bf16)v;
  }
  __syncthreads();
  int wave = tid >> 5, lane = tid & 31;
  for (int nt = wave; nt < HID / 16; nt += 8) {
    f32x8 acc = wmma_row(a, IN_PAD, Wp, HID / 16, nt, IN_PAD / 32);
    int n = lane & 15, mh = lane >> 4;
    float b = bias[nt * 16 + n];
#pragma unroll
    for (int r = 0; r < 8; ++r)
      h[(size_t)(tile * 16 + mh * 8 + r) * HID + nt * 16 + n] = acc[r] + b;
  }
}

// ---------------- FiLM: h = s*h + b (in place), also produce bf16 copy ----------------
__global__ void k_film(float* __restrict__ h, bf16* __restrict__ hb,
                       const float* __restrict__ film, int l) {
  size_t i = (size_t)blockIdx.x * blockDim.x + threadIdx.x;
  if (i >= (size_t)N_NODES * HID) return;
  int c = (int)(i & 255);
  float s = film[l * 2 * HID + c];
  float b = film[l * 2 * HID + HID + c];
  float v = s * h[i] + b;
  h[i] = v;
  hb[i] = (bf16)v;
}

// ---------------- fused per-edge MLP (one layer) ----------------
// 16 edges / block, 8 waves. Gathers h[row], h[col] via async-to-LDS (no VGPR
// round trip), builds m = [h[row], h[col], radial, h_e] (800 wide, zero padded),
// chains three WMMA GEMMs in LDS, then c_w2 dot + atomic segment sums.
__global__ __launch_bounds__(256) void k_edge(
    const int* __restrict__ ei, const bf16* __restrict__ hb,
    const float* __restrict__ ea, const float* __restrict__ ew,
    const float* __restrict__ ebias, const float* __restrict__ coord,
    const bf16* __restrict__ W1p, const float* __restrict__ b1,
    const bf16* __restrict__ W2p, const float* __restrict__ b2,
    const bf16* __restrict__ Wc1p, const float* __restrict__ bc1,
    const float* __restrict__ cw2,
    float* __restrict__ agg, float* __restrict__ trans) {
  __shared__ __align__(16) bf16 m[16 * M_PAD];     // 25.6 KB, reused for ef & t1
  __shared__ __align__(16) bf16 t2[16 * HID];      // 8 KB
  __shared__ float cdiff[16][3];
  __shared__ int rowi[16], coli[16];
  const int tid = threadIdx.x;
  const int e0 = blockIdx.x * 16;

  if (tid < 16) {
    int r = ei[e0 + tid];
    int c = ei[N_EDGES + e0 + tid];
    rowi[tid] = r; coli[tid] = c;
    float dx = coord[r * 3 + 0] - coord[c * 3 + 0];
    float dy = coord[r * 3 + 1] - coord[c * 3 + 1];
    float dz = coord[r * 3 + 2] - coord[c * 3 + 2];
    float radial = dx * dx + dy * dy + dz * dz;
    float inv = 1.0f / (sqrtf(radial) + EPS_F);
    cdiff[tid][0] = dx * inv; cdiff[tid][1] = dy * inv; cdiff[tid][2] = dz * inv;
    m[tid * M_PAD + 512] = (bf16)radial;
    for (int c2 = M_CAT; c2 < M_PAD; ++c2) m[tid * M_PAD + c2] = (bf16)0.0f;
  }
  __syncthreads();

  // gather: async memory->LDS, 512B (32 lanes x 16B) per instruction.
  // 16 edges x 2 segments = 32 jobs, 4 per wave.
  {
    int lane = tid & 31, wv = tid >> 5;
#pragma unroll
    for (int j = 0; j < 4; ++j) {
      int job = wv * 4 + j;
      int e = job >> 1, seg = job & 1;
      int node = seg ? coli[e] : rowi[e];
      unsigned long long g =
          (unsigned long long)(uintptr_t)(hb + (size_t)node * HID) + (unsigned)(lane * 16);
      unsigned l32 = lds_addr32(m + e * M_PAD + seg * 256) + (unsigned)(lane * 16);
      async_copy_b128(l32, g);
    }
    // h_e recomputed on the fly (rank-1): 16 threads per edge, 16 cols each
    int le = tid >> 4, sub = tid & 15;
    float eav = ea[e0 + le];
    for (int i = 0; i < 16; ++i) {
      int c = sub * 16 + i;
      m[le * M_PAD + 513 + c] = (bf16)(eav * ew[c] + ebias[c]);
    }
    wait_asynccnt0();
  }
  __syncthreads();

  int wave = tid >> 5, lane = tid & 31;
  // stage 1: silu(m @ e_w1 + e_b1) -> t2
  for (int nt = wave; nt < 16; nt += 8) {
    f32x8 acc = wmma_row(m, M_PAD, W1p, 16, nt, M_PAD / 32);
    int n = lane & 15, mh = lane >> 4;
    float b = b1[nt * 16 + n];
#pragma unroll
    for (int r = 0; r < 8; ++r)
      t2[(mh * 8 + r) * HID + nt * 16 + n] = (bf16)silu_f(acc[r] + b);
  }
  __syncthreads();

  // stage 2: edge_feat = silu(t2 @ e_w2 + e_b2) -> overlay m[0..4096)
  bf16* ef = m;
  for (int nt = wave; nt < 16; nt += 8) {
    f32x8 acc = wmma_row(t2, HID, W2p, 16, nt, HID / 32);
    int n = lane & 15, mh = lane >> 4;
    float b = b2[nt * 16 + n];
#pragma unroll
    for (int r = 0; r < 8; ++r)
      ef[(mh * 8 + r) * HID + nt * 16 + n] = (bf16)silu_f(acc[r] + b);
  }
  __syncthreads();

  // agg = segment_sum(edge_feat, row)
  for (int i = tid; i < 16 * HID; i += 256) {
    int e = i >> 8, c = i & 255;
    atomicAdd(&agg[(size_t)rowi[e] * HID + c], (float)ef[i]);
  }

  // stage 3: t1 = silu(ef @ c_w1 + c_b1) -> overlay m[4096..8192)
  bf16* t1 = m + 16 * HID;
  for (int nt = wave; nt < 16; nt += 8) {
    f32x8 acc = wmma_row(ef, HID, Wc1p, 16, nt, HID / 32);
    int n = lane & 15, mh = lane >> 4;
    float b = bc1[nt * 16 + n];
#pragma unroll
    for (int r = 0; r < 8; ++r)
      t1[(mh * 8 + r) * HID + nt * 16 + n] = (bf16)silu_f(acc[r] + b);
  }
  __syncthreads();

  // stage 4: s = t1 . c_w2 (per edge), trans = cdiff * s, atomic segment-sum
  {
    int e = tid >> 4, part = tid & 15;
    float p = 0.0f;
    for (int i = 0; i < 16; ++i) {
      int c = part * 16 + i;
      p += (float)t1[e * HID + c] * cw2[c];
    }
    for (int o = 8; o > 0; o >>= 1) p += __shfl_down(p, o, 16);
    if (part == 0) {
      atomicAdd(&trans[(size_t)rowi[e] * 3 + 0], cdiff[e][0] * p);
      atomicAdd(&trans[(size_t)rowi[e] * 3 + 1], cdiff[e][1] * p);
      atomicAdd(&trans[(size_t)rowi[e] * 3 + 2], cdiff[e][2] * p);
    }
  }
}

// ---------------- node MLP: h += silu([h,agg]@n_w1+b1) @ n_w2 + b2 ----------------
__global__ __launch_bounds__(256) void k_node_mlp(
    const bf16* __restrict__ hb, const float* __restrict__ agg,
    const bf16* __restrict__ W1p, const float* __restrict__ b1,
    const bf16* __restrict__ W2p, const float* __restrict__ b2,
    float* __restrict__ h) {
  __shared__ __align__(16) bf16 cat[16 * 512];
  __shared__ __align__(16) bf16 u[16 * HID];
  int tid = threadIdx.x, n0 = blockIdx.x * 16;
  // first half of cat (h, already bf16): async memory->LDS; 16 rows, 2 per wave
  {
    int lane = tid & 31, wv = tid >> 5;
#pragma unroll
    for (int j = 0; j < 2; ++j) {
      int r = wv * 2 + j;
      unsigned long long g =
          (unsigned long long)(uintptr_t)(hb + (size_t)(n0 + r) * HID) + (unsigned)(lane * 16);
      unsigned l32 = lds_addr32(cat + r * 512) + (unsigned)(lane * 16);
      async_copy_b128(l32, g);
    }
    // second half (agg, f32 -> bf16 conversion) on VALU
    for (int i = tid; i < 16 * HID; i += 256) {
      int r = i >> 8, c = i & 255;
      cat[r * 512 + 256 + c] = (bf16)agg[(size_t)(n0 + r) * HID + c];
    }
    wait_asynccnt0();
  }
  __syncthreads();
  int wave = tid >> 5, lane = tid & 31;
  for (int nt = wave; nt < 16; nt += 8) {
    f32x8 acc = wmma_row(cat, 512, W1p, 16, nt, 512 / 32);
    int n = lane & 15, mh = lane >> 4;
    float b = b1[nt * 16 + n];
#pragma unroll
    for (int r = 0; r < 8; ++r)
      u[(mh * 8 + r) * HID + nt * 16 + n] = (bf16)silu_f(acc[r] + b);
  }
  __syncthreads();
  for (int nt = wave; nt < 16; nt += 8) {
    f32x8 acc = wmma_row(u, HID, W2p, 16, nt, HID / 32);
    int n = lane & 15, mh = lane >> 4;
    float b = b2[nt * 16 + n];
#pragma unroll
    for (int r = 0; r < 8; ++r) {
      size_t idx = (size_t)(n0 + mh * 8 + r) * HID + nt * 16 + n;
      h[idx] = h[idx] + acc[r] + b;
    }
  }
}

// ---------------- misc elementwise / segment kernels ----------------
__global__ void k_cnt(const int* __restrict__ ei, float* __restrict__ cnt) {
  int e = blockIdx.x * blockDim.x + threadIdx.x;
  if (e < N_EDGES) atomicAdd(&cnt[ei[e]], 1.0f);
}

__global__ void k_coord(float* __restrict__ coord, const float* __restrict__ trans,
                        const float* __restrict__ cnt) {
  int n = blockIdx.x * blockDim.x + threadIdx.x;
  if (n < N_NODES) {
    float c = cnt[n]; if (c < 1.0f) c = 1.0f;
    coord[n * 3 + 0] += trans[n * 3 + 0] / c;
    coord[n * 3 + 1] += trans[n * 3 + 1] / c;
    coord[n * 3 + 2] += trans[n * 3 + 2] / c;
  }
}

// batch is sorted -> accumulate runs in registers, one atomic per group change.
__global__ __launch_bounds__(256) void k_gsum(const float* __restrict__ h,
                                              const int* __restrict__ batch,
                                              float* __restrict__ gsum) {
  __shared__ int bg[GS_NODES];
  int c = threadIdx.x;                 // 256 columns
  int n0 = blockIdx.x * GS_NODES;
  int cnt = N_NODES - n0; if (cnt > GS_NODES) cnt = GS_NODES;
  if (c < cnt) bg[c] = batch[n0 + c];
  __syncthreads();
  float run = 0.0f;
  int curg = bg[0];
  for (int i = 0; i < cnt; ++i) {
    int g = bg[i];
    if (g != curg) {
      atomicAdd(&gsum[(size_t)curg * HID + c], run);
      run = 0.0f; curg = g;
    }
    run += h[(size_t)(n0 + i) * HID + c];
  }
  atomicAdd(&gsum[(size_t)curg * HID + c], run);
}

__global__ void k_gcnt(const int* __restrict__ batch, float* __restrict__ gcnt) {
  int n = blockIdx.x * blockDim.x + threadIdx.x;
  if (n < N_NODES) atomicAdd(&gcnt[batch[n]], 1.0f);
}

__global__ void k_copy_coord(const float* __restrict__ coord, float* __restrict__ out) {
  int i = blockIdx.x * blockDim.x + threadIdx.x;
  if (i < N_NODES * 3) out[(size_t)N_NODES * IN_DIM + i] = coord[i];
}

// ---------------- prediction head ----------------
__global__ __launch_bounds__(256) void k_pred(
    const float* __restrict__ h, const float* __restrict__ gsum,
    const float* __restrict__ gcnt, const int* __restrict__ batch,
    const bf16* __restrict__ W1p, const float* __restrict__ b1,
    const bf16* __restrict__ W2p, const float* __restrict__ b2,
    const bf16* __restrict__ W3p, const float* __restrict__ b3,
    float* __restrict__ out) {
  __shared__ __align__(16) bf16 cat[16 * 512];
  __shared__ __align__(16) bf16 z1[16 * HID];
  __shared__ __align__(16) bf16 z2[16 * HID];
  int tid = threadIdx.x, n0 = blockIdx.x * 16;
  for (int i = tid; i < 16 * 512; i += 256) {
    int r = i >> 9, c = i & 511;
    float v;
    if (c < HID) {
      int b = batch[n0 + r];
      float cc = gcnt[b]; if (cc < 1.0f) cc = 1.0f;
      v = gsum[(size_t)b * HID + c] / cc;
    } else {
      v = h[(size_t)(n0 + r) * HID + (c - HID)];
    }
    cat[i] = (bf16)v;
  }
  __syncthreads();
  int wave = tid >> 5, lane = tid & 31;
  for (int nt = wave; nt < 16; nt += 8) {
    f32x8 acc = wmma_row(cat, 512, W1p, 16, nt, 16);
    int n = lane & 15, mh = lane >> 4;
    float b = b1[nt * 16 + n];
#pragma unroll
    for (int r = 0; r < 8; ++r) {
      float v = acc[r] + b;
      z1[(mh * 8 + r) * HID + nt * 16 + n] = (bf16)(v > 0.0f ? v : 0.0f);
    }
  }
  __syncthreads();
  for (int nt = wave; nt < 16; nt += 8) {
    f32x8 acc = wmma_row(z1, HID, W2p, 16, nt, 8);
    int n = lane & 15, mh = lane >> 4;
    float b = b2[nt * 16 + n];
#pragma unroll
    for (int r = 0; r < 8; ++r) {
      float v = acc[r] + b;
      z2[(mh * 8 + r) * HID + nt * 16 + n] = (bf16)(v > 0.0f ? v : 0.0f);
    }
  }
  __syncthreads();
  for (int nt = wave; nt < IN_DIM / 16; nt += 8) {
    f32x8 acc = wmma_row(z2, HID, W3p, IN_DIM / 16, nt, 8);
    int n = lane & 15, mh = lane >> 4;
    float b = b3[nt * 16 + n];
#pragma unroll
    for (int r = 0; r < 8; ++r)
      out[(size_t)(n0 + mh * 8 + r) * IN_DIM + nt * 16 + n] = acc[r] + b;
  }
}

// ---------------- host launch ----------------
extern "C" void kernel_launch(void* const* d_in, const int* in_sizes, int n_in,
                              void* d_out, int out_size, void* d_ws, size_t ws_size,
                              hipStream_t stream) {
  (void)in_sizes; (void)n_in; (void)out_size; (void)ws_size;
  const float* x      = (const float*)d_in[0];
  const int*   ei     = (const int*)d_in[1];
  const float* ea     = (const float*)d_in[2];
  const float* xcoord = (const float*)d_in[3];
  const float* film   = (const float*)d_in[4];
  const int*   batch  = (const int*)d_in[5];
  const float* node_w = (const float*)d_in[6];
  const float* node_b = (const float*)d_in[7];
  const float* edge_w = (const float*)d_in[8];
  const float* edge_b = (const float*)d_in[9];
  const float* pred_w1 = (const float*)d_in[65];
  const float* pred_b1 = (const float*)d_in[66];
  const float* pred_w2 = (const float*)d_in[67];
  const float* pred_b2 = (const float*)d_in[68];
  const float* pred_w3 = (const float*)d_in[69];
  const float* pred_b3 = (const float*)d_in[70];
  float* out = (float*)d_out;

  // workspace carving
  char* w = (char*)d_ws;
  auto alloc = [&](size_t bytes) {
    void* p = (void*)w;
    w += (bytes + 255) & ~(size_t)255;
    return p;
  };
  float* d_h     = (float*)alloc((size_t)N_NODES * HID * 4);
  bf16*  d_hbf   = (bf16*) alloc((size_t)N_NODES * HID * 2);
  float* d_agg   = (float*)alloc((size_t)N_NODES * HID * 4);
  float* d_trans = (float*)alloc((size_t)N_NODES * 3 * 4);
  float* d_cntr  = (float*)alloc((size_t)N_NODES * 4);
  float* d_coord = (float*)alloc((size_t)N_NODES * 3 * 4);
  float* d_gsum  = (float*)alloc((size_t)NGROUP * HID * 4);
  float* d_gcnt  = (float*)alloc((size_t)NGROUP * 4);
  bf16* p_nodew = (bf16*)alloc((size_t)IN_PAD * HID * 2);
  bf16 *p_ew1[NLAYERS], *p_ew2[NLAYERS], *p_nw1[NLAYERS], *p_nw2[NLAYERS], *p_cw1[NLAYERS];
  for (int l = 0; l < NLAYERS; ++l) {
    p_ew1[l] = (bf16*)alloc((size_t)M_PAD * HID * 2);
    p_ew2[l] = (bf16*)alloc((size_t)HID * HID * 2);
    p_nw1[l] = (bf16*)alloc((size_t)512 * HID * 2);
    p_nw2[l] = (bf16*)alloc((size_t)HID * HID * 2);
    p_cw1[l] = (bf16*)alloc((size_t)HID * HID * 2);
  }
  bf16* p_pw1 = (bf16*)alloc((size_t)512 * HID * 2);
  bf16* p_pw2 = (bf16*)alloc((size_t)HID * HID * 2);
  bf16* p_pw3 = (bf16*)alloc((size_t)HID * IN_DIM * 2);

  auto repack = [&](const float* W, bf16* Wp, int Ksrc, int Kpad, int Nn) {
    int total = Kpad * Nn;
    k_repack<<<(total + 255) / 256, 256, 0, stream>>>(W, Wp, Ksrc, Kpad, Nn);
  };
  repack(node_w, p_nodew, IN_DIM, IN_PAD, HID);
  for (int l = 0; l < NLAYERS; ++l) {
    int base = 10 + 11 * l;
    repack((const float*)d_in[base + 0], p_ew1[l], M_CAT, M_PAD, HID);
    repack((const float*)d_in[base + 2], p_ew2[l], HID, HID, HID);
    repack((const float*)d_in[base + 4], p_nw1[l], 512, 512, HID);
    repack((const float*)d_in[base + 6], p_nw2[l], HID, HID, HID);
    repack((const float*)d_in[base + 8], p_cw1[l], HID, HID, HID);
  }
  repack(pred_w1, p_pw1, 512, 512, HID);
  repack(pred_w2, p_pw2, HID, HID, HID);
  repack(pred_w3, p_pw3, HID, HID, IN_DIM);

  hipMemsetAsync(d_cntr, 0, (size_t)N_NODES * 4, stream);
  k_cnt<<<(N_EDGES + 255) / 256, 256, 0, stream>>>(ei, d_cntr);
  hipMemcpyAsync(d_coord, xcoord, (size_t)N_NODES * 3 * 4, hipMemcpyDeviceToDevice, stream);

  k_node_embed<<<N_NODES / 16, 256, 0, stream>>>(x, p_nodew, node_b, d_h);

  for (int l = 0; l < NLAYERS; ++l) {
    int base = 10 + 11 * l;
    const float* b_e1 = (const float*)d_in[base + 1];
    const float* b_e2 = (const float*)d_in[base + 3];
    const float* b_n1 = (const float*)d_in[base + 5];
    const float* b_n2 = (const float*)d_in[base + 7];
    const float* b_c1 = (const float*)d_in[base + 9];
    const float* cw2  = (const float*)d_in[base + 10];

    k_film<<<(N_NODES * HID) / 256, 256, 0, stream>>>(d_h, d_hbf, film, l);
    hipMemsetAsync(d_agg, 0, (size_t)N_NODES * HID * 4, stream);
    hipMemsetAsync(d_trans, 0, (size_t)N_NODES * 3 * 4, stream);
    k_edge<<<N_EDGES / 16, 256, 0, stream>>>(ei, d_hbf, ea, edge_w, edge_b, d_coord,
                                             p_ew1[l], b_e1, p_ew2[l], b_e2,
                                             p_cw1[l], b_c1, cw2, d_agg, d_trans);
    k_coord<<<(N_NODES + 255) / 256, 256, 0, stream>>>(d_coord, d_trans, d_cntr);
    k_node_mlp<<<N_NODES / 16, 256, 0, stream>>>(d_hbf, d_agg, p_nw1[l], b_n1,
                                                 p_nw2[l], b_n2, d_h);
  }

  hipMemsetAsync(d_gsum, 0, (size_t)NGROUP * HID * 4, stream);
  hipMemsetAsync(d_gcnt, 0, (size_t)NGROUP * 4, stream);
  k_gsum<<<(N_NODES + GS_NODES - 1) / GS_NODES, 256, 0, stream>>>(d_h, batch, d_gsum);
  k_gcnt<<<(N_NODES + 255) / 256, 256, 0, stream>>>(batch, d_gcnt);
  k_pred<<<N_NODES / 16, 256, 0, stream>>>(d_h, d_gsum, d_gcnt, batch,
                                           p_pw1, pred_b1, p_pw2, pred_b2,
                                           p_pw3, pred_b3, out);
  k_copy_coord<<<(N_NODES * 3 + 255) / 256, 256, 0, stream>>>(d_coord, out);
}
